// OptimizedSparseSimilarity_80135499809313
// MI455X (gfx1250) — compile-verified, matrix operand order checked by reading, not run on hardware
//
#include <hip/hip_runtime.h>
#include <hip/hip_bf16.h>

#define TAU_INV 5.0f
#define KNN 15
#define BATCH 2
#define NX 8192
#define NY 8192
#define CDIM 256

typedef _Float16 half_t;
typedef __attribute__((ext_vector_type(16))) _Float16 v16h;
typedef __attribute__((ext_vector_type(8)))  _Float16 v8h;
typedef __attribute__((ext_vector_type(8)))  float    v8f;
typedef __attribute__((ext_vector_type(4)))  float    v4f;

union Frag {
    v16h v;
    v4f  f[2];   // two 16-byte chunks (8 halves each)
};

// ---------------------------------------------------------------------------
// Kernel 1: L2-normalize rows of feat_x / feat_y and convert to f16.
// One wave32 per row (C = 256 -> 8 floats per lane).
// ---------------------------------------------------------------------------
__global__ __launch_bounds__(256)
void normalize_to_f16_kernel(const float* __restrict__ fx,
                             const float* __restrict__ fy,
                             half_t* __restrict__ fxh,
                             half_t* __restrict__ fyh)
{
    const int wave = (int)((blockIdx.x * blockDim.x + threadIdx.x) >> 5);
    const int lane = (int)(threadIdx.x & 31);
    const int rowsX = BATCH * NX;

    const float* src;
    half_t* dst;
    long row;
    if (wave < rowsX) { src = fx; dst = fxh; row = wave; }
    else              { src = fy; dst = fyh; row = wave - rowsX; }

    const float* p = src + row * (long)CDIM + lane * 8;
    v4f a = *(const v4f*)p;
    v4f b = *(const v4f*)(p + 4);

    float ss = 0.0f;
#pragma unroll
    for (int i = 0; i < 4; ++i) ss += a[i] * a[i] + b[i] * b[i];
#pragma unroll
    for (int m = 16; m >= 1; m >>= 1) ss += __shfl_xor(ss, m, 32);

    const float norm  = sqrtf(ss);
    const float scale = 1.0f / fmaxf(norm, 1e-12f);

    v8h o;
#pragma unroll
    for (int i = 0; i < 4; ++i) {
        o[i]     = (half_t)(a[i] * scale);
        o[i + 4] = (half_t)(b[i] * scale);
    }
    *(v8h*)(dst + row * (long)CDIM + lane * 8) = o;
}

// ---------------------------------------------------------------------------
// Kernel 2: fused WMMA GEMM (sim = fx @ fy^T) + per-row top-15 + softmax.
// 4 waves per block; each wave owns a 16-row strip of X for one batch and
// sweeps all Ny columns in 16-wide tiles, 16x16x32 f16 WMMA over C=256.
// ---------------------------------------------------------------------------
__global__ __launch_bounds__(128)
void simtopk_kernel(const half_t* __restrict__ fxh,
                    const half_t* __restrict__ fyh,
                    float* __restrict__ out_vals,
                    float* __restrict__ out_idx)
{
    __shared__ float lds[4][16 * 17];   // per-wave 16x16 tile, stride 17 (bank-conflict free)

    const int wv    = (int)(threadIdx.x >> 5);   // wave in block: 0..3
    const int lane  = (int)(threadIdx.x & 31);
    const int rowTile = (int)blockIdx.x * 4 + wv;        // 0 .. B*Nx/16 - 1
    const int b     = rowTile / (NX / 16);
    const int rbase = (rowTile % (NX / 16)) * 16;

    const int mrow = lane & 15;            // A row / B column handled by this lane
    const int koff = (lane >> 4) * 8;      // K sub-offset per ISA 16-bit A/B layout

    // --- preload A fragments: 16 rows x 256 K, 8 chunks of K=32, reused 512x ---
    Frag A[8];
    {
        const half_t* arow = fxh + ((long)b * NX + rbase + mrow) * CDIM + koff;
#pragma unroll
        for (int kc = 0; kc < 8; ++kc) {
            const half_t* p = arow + kc * 32;
            A[kc].f[0] = *(const v4f*)p;          // K = koff .. koff+7
            A[kc].f[1] = *(const v4f*)(p + 16);   // K = koff+16 .. koff+23
        }
    }

    // --- per-row top-k state (lanes 0..15 each own one row), register resident ---
    float kvals[KNN];
    int   kidx[KNN];
#pragma unroll
    for (int j = 0; j < KNN; ++j) { kvals[j] = -3.0e38f; kidx[j] = 0; }

    float* tile = &lds[wv][0];
    const half_t* bbat = fyh + (long)b * NY * CDIM;

    for (int nt = 0; nt < NY / 16; ++nt) {
        v8f acc = {};
        const half_t* brow = bbat + ((long)(nt * 16 + mrow)) * CDIM + koff;
#pragma unroll
        for (int kc = 0; kc < 8; ++kc) {
            Frag Bf;
            const half_t* p = brow + kc * 32;
            Bf.f[0] = *(const v4f*)p;
            Bf.f[1] = *(const v4f*)(p + 16);
            acc = __builtin_amdgcn_wmma_f32_16x16x32_f16(
                false, A[kc].v, false, Bf.v, (short)0, acc, false, false);
        }

        // spill 16x16 C tile to LDS: VGPR v -> row (v + 8*(lane>=16)), col = lane&15
#pragma unroll
        for (int v = 0; v < 8; ++v) {
            tile[(v + ((lane >> 4) * 8)) * 17 + (lane & 15)] = acc[v];
        }
        asm volatile("s_wait_dscnt 0" ::: "memory");
        __builtin_amdgcn_wave_barrier();

        if (lane < 16) {
#pragma unroll
            for (int c = 0; c < 16; ++c) {
                float v = tile[lane * 17 + c];
                if (v > kvals[0]) {                       // beats current min?
                    float cv = v; int ci = nt * 16 + c;   // bubble-insert, fully unrolled
#pragma unroll
                    for (int j = 0; j < KNN - 1; ++j) {
                        bool  mv = cv > kvals[j + 1];
                        float sv = mv ? kvals[j + 1] : cv;
                        int   si = mv ? kidx[j + 1]  : ci;
                        float nc = mv ? cv : kvals[j + 1];
                        int   ni = mv ? ci : kidx[j + 1];
                        kvals[j] = sv; kidx[j] = si; cv = nc; ci = ni;
                    }
                    kvals[KNN - 1] = cv; kidx[KNN - 1] = ci;
                }
            }
        }
        __builtin_amdgcn_wave_barrier();
    }

    // --- softmax over kept values (scaled by 1/tau), emit in descending order ---
    if (lane < 16) {
        const long row = (long)b * NX + rbase + lane;
        const float mx = kvals[KNN - 1] * TAU_INV;
        float e[KNN];
        float s = 0.0f;
#pragma unroll
        for (int j = 0; j < KNN; ++j) { e[j] = __expf(kvals[j] * TAU_INV - mx); s += e[j]; }
        const float inv = 1.0f / s;
        float* ov = out_vals + row * KNN;
        float* oi = out_idx  + row * KNN;
#pragma unroll
        for (int t = 0; t < KNN; ++t) {
            ov[t] = e[KNN - 1 - t] * inv;
            oi[t] = (float)kidx[KNN - 1 - t];
        }
    }
}

extern "C" void kernel_launch(void* const* d_in, const int* in_sizes, int n_in,
                              void* d_out, int out_size, void* d_ws, size_t ws_size,
                              hipStream_t stream)
{
    (void)in_sizes; (void)n_in; (void)out_size; (void)ws_size;

    const float* fx = (const float*)d_in[0];
    const float* fy = (const float*)d_in[1];

    half_t* fxh = (half_t*)d_ws;
    half_t* fyh = fxh + (size_t)BATCH * NX * CDIM;

    float* ov = (float*)d_out;
    float* oi = ov + (size_t)BATCH * NX * KNN;

    // Pass 1: normalize + f32 -> f16 (one wave per row, 8 rows per 256-thread block)
    const int totalRows = BATCH * (NX + NY);
    normalize_to_f16_kernel<<<totalRows / 8, 256, 0, stream>>>(fx, fy, fxh, fyh);

    // Pass 2: fused WMMA GEMM + top-k + softmax (4 waves/block, 64 rows per block)
    const int gblocks = (BATCH * NX) / 64;
    simtopk_kernel<<<gblocks, 128, 0, stream>>>(fxh, fyh, ov, oi);
}
